// M3GBlock_7550552507075
// MI455X (gfx1250) — matrix operand order repaired and checked by reading, not compiled, not run on hardware
//
#include <hip/hip_runtime.h>

#define DIM    128
#define NRBF   64
#define NNODES 100000
#define NEDGES 640000

typedef __attribute__((ext_vector_type(16))) _Float16 v16h;
typedef __attribute__((ext_vector_type(8)))  _Float16 v8h;
typedef __attribute__((ext_vector_type(4)))  _Float16 v4h;
typedef __attribute__((ext_vector_type(8)))  float    v8f;

// ---- fragment table bases (each fragment = 32 lanes * 16 halves = 512 halves) ----
#define EW1_BASE 0    // K=320 -> 10 ktiles * 8 ntiles
#define EW2_BASE 80   // K=128 -> 4*8
#define TW1_BASE 112  // K=129 padded to 160 -> 5*8
#define TW2_BASE 152  // 4*8
#define NW1_BASE 184  // K=256 -> 8*8
#define NW2_BASE 248  // 4*8
#define NFRAGS   280

// ---- workspace layout (bytes) ----
#define WFRAG_BYTES ((size_t)NFRAGS * 512 * 2)              // 286,720
#define VEC_OFF     WFRAG_BYTES                              // vec_sum: N*3 f32
#define CNT_OFF     (VEC_OFF + (size_t)NNODES * 3 * 4)       // counts:  N f32
#define NVS_OFF     (CNT_OFF + (size_t)NNODES * 4)           // normalized: N*3 f32

__device__ __forceinline__ float swishf(float v) { return v / (1.0f + __expf(-v)); }
__device__ __forceinline__ float sigmf(float v)  { return 1.0f / (1.0f + __expf(-v)); }

// A-matrix 16x32 f16 fragment from row-major f16 LDS tile.
// ISA layout: lane<16 -> halves 0..7 = K base+0..7, halves 8..15 = K base+16..23
//             lane>=16 -> same +8.  M = lane & 15.
__device__ __forceinline__ v16h load_A_lds(const _Float16* rowBase, int rowStride,
                                           int kt, int lane) {
  int m    = lane & 15;
  int koff = kt * 32 + ((lane >> 4) << 3);
  const _Float16* p = rowBase + m * rowStride + koff;
  v8h lo = *(const v8h*)p;
  v8h hi = *(const v8h*)(p + 16);
  v16h a;
#pragma unroll
  for (int i = 0; i < 8; ++i) { a[i] = lo[i]; a[i + 8] = hi[i]; }
  return a;
}

// B fragment pre-packed in workspace: one coalesced 32B load per lane.
__device__ __forceinline__ v16h load_B(const _Float16* __restrict__ wfrag, int frag, int lane) {
  return *(const v16h*)(wfrag + (size_t)frag * 512 + lane * 16);
}

#define WMMA_F16(A, B, C) \
  __builtin_amdgcn_wmma_f32_16x16x32_f16(false, (A), false, (B), (short)0, (C), false, false)

// ---------------- weight prep: pack W[k][n] (row-major f32) into B fragments ----
__global__ void prep_weights(const float* __restrict__ ew1, const float* __restrict__ ew2,
                             const float* __restrict__ tw1, const float* __restrict__ tw2,
                             const float* __restrict__ nw1, const float* __restrict__ nw2,
                             _Float16* __restrict__ wfrag) {
  int f = blockIdx.x;
  int lane = threadIdx.x;
  const float* W = ew1; int Krows = 320; int fl = f;
  if (f >= 248)      { W = nw2; Krows = 128; fl = f - 248; }
  else if (f >= 184) { W = nw1; Krows = 256; fl = f - 184; }
  else if (f >= 152) { W = tw2; Krows = 128; fl = f - 152; }
  else if (f >= 112) { W = tw1; Krows = 129; fl = f - 112; }
  else if (f >= 80)  { W = ew2; Krows = 128; fl = f - 80; }
  int kt = fl >> 3, nt = fl & 7;
  int n = nt * 16 + (lane & 15);
  _Float16* dst = wfrag + (size_t)f * 512 + lane * 16;
#pragma unroll
  for (int h = 0; h < 16; ++h) {
    int k = kt * 32 + ((lane >> 4) << 3) + (h & 7) + ((h >> 3) << 4);   // lanes 0-15: K 0..7,16..23; lanes 16-31: K 8..15,24..31
    dst[h] = (k < Krows) ? (_Float16)W[(size_t)k * DIM + n] : (_Float16)0.0f;
  }
}

// ---------------- scatter edge_vec sums + counts ----------------
__global__ void scatter_vec(const int* __restrict__ coli, const float* __restrict__ evec,
                            float* __restrict__ vsum, float* __restrict__ cnt) {
  int e = blockIdx.x * 256 + threadIdx.x;
  if (e >= NEDGES) return;
  int c = coli[e];
  atomicAdd(vsum + (size_t)c * 3 + 0, evec[(size_t)e * 3 + 0]);
  atomicAdd(vsum + (size_t)c * 3 + 1, evec[(size_t)e * 3 + 1]);
  atomicAdd(vsum + (size_t)c * 3 + 2, evec[(size_t)e * 3 + 2]);
  atomicAdd(cnt + c, 1.0f);
}

__global__ void normalize_vec(const float* __restrict__ vsum, const float* __restrict__ cnt,
                              float* __restrict__ nvs) {
  int i = blockIdx.x * 256 + threadIdx.x;
  if (i >= NNODES) return;
  float d = fmaxf(cnt[i], 1.0f);
  nvs[(size_t)i * 3 + 0] = vsum[(size_t)i * 3 + 0] / d;
  nvs[(size_t)i * 3 + 1] = vsum[(size_t)i * 3 + 1] / d;
  nvs[(size_t)i * 3 + 2] = vsum[(size_t)i * 3 + 2] / d;
}

// ---------------- fused edge pipeline: 4 GEMM layers + gate + scatter ----------
// 128 threads = 4 waves; each wave owns 16 edges (wave-private LDS tiles,
// LDS ops are in-order per wave -> no block barriers needed).
__global__ __launch_bounds__(128) void edge_kernel(
    const float* __restrict__ x, const int* __restrict__ rowi, const int* __restrict__ coli,
    const float* __restrict__ eattr, const float* __restrict__ evec,
    const float* __restrict__ elen, const float* __restrict__ nvs,
    const _Float16* __restrict__ wfrag,
    const float* __restrict__ eb1, const float* __restrict__ eb2,
    const float* __restrict__ tb1, const float* __restrict__ tb2,
    float* __restrict__ aggr) {
  __shared__ _Float16 s_in[64 * 320];   // input staging; later reused as g_in[0..159]
  __shared__ _Float16 s_h1[64 * 128];   // hidden staging (h1, later t1)

  int lane  = threadIdx.x & 31;
  int wave  = threadIdx.x >> 5;
  int tile0 = wave * 16;                     // wave's first LDS row
  int e0    = blockIdx.x * 64 + tile0;       // wave's first global edge
  int n     = lane & 15;
  int rb    = (lane >> 4) * 8;

  // ---- stage [x[row] | edge_attr | x[col]] as f16 ----
#pragma unroll 1
  for (int e = 0; e < 16; ++e) {
    int eg = e0 + e;
    int r = rowi[eg], c = coli[eg];
    float4 xr = *(const float4*)(x + (size_t)r * DIM + lane * 4);
    float2 ea = *(const float2*)(eattr + (size_t)eg * NRBF + lane * 2);
    float4 xc = *(const float4*)(x + (size_t)c * DIM + lane * 4);
    _Float16* rw = s_in + (tile0 + e) * 320;
    v4h h;
    h[0] = (_Float16)xr.x; h[1] = (_Float16)xr.y; h[2] = (_Float16)xr.z; h[3] = (_Float16)xr.w;
    *(v4h*)(rw + lane * 4) = h;
    rw[128 + lane * 2]     = (_Float16)ea.x;
    rw[128 + lane * 2 + 1] = (_Float16)ea.y;
    h[0] = (_Float16)xc.x; h[1] = (_Float16)xc.y; h[2] = (_Float16)xc.z; h[3] = (_Float16)xc.w;
    *(v4h*)(rw + 192 + lane * 4) = h;
  }

  // ---- Layer 1: 320 -> 128, swish ----
#pragma unroll
  for (int nt = 0; nt < 8; ++nt) {
    v8f acc = {};
#pragma unroll
    for (int kt = 0; kt < 10; ++kt) {
      v16h a = load_A_lds(s_in + tile0 * 320, 320, kt, lane);
      v16h b = load_B(wfrag, EW1_BASE + kt * 8 + nt, lane);
      acc = WMMA_F16(a, b, acc);
    }
    float bn = eb1[nt * 16 + n];
#pragma unroll
    for (int r = 0; r < 8; ++r)
      s_h1[(tile0 + rb + r) * 128 + nt * 16 + n] = (_Float16)swishf(acc[r] + bn);
  }

  // ---- Layer 2: 128 -> 128, swish -> m_ij, stored into g_in[0..127] (reuses s_in) ----
#pragma unroll
  for (int nt = 0; nt < 8; ++nt) {
    v8f acc = {};
#pragma unroll
    for (int kt = 0; kt < 4; ++kt) {
      v16h a = load_A_lds(s_h1 + tile0 * 128, 128, kt, lane);
      v16h b = load_B(wfrag, EW2_BASE + kt * 8 + nt, lane);
      acc = WMMA_F16(a, b, acc);
    }
    float bn = eb2[nt * 16 + n];
#pragma unroll
    for (int r = 0; r < 8; ++r)
      s_in[(tile0 + rb + r) * 320 + nt * 16 + n] = (_Float16)swishf(acc[r] + bn);
  }

  // ---- cos(theta) + zero pad into g_in[128..159] ----
  if (lane < 16) {
    int eg = e0 + lane;
    int c = coli[eg];
    float vx = evec[(size_t)eg * 3], vy = evec[(size_t)eg * 3 + 1], vz = evec[(size_t)eg * 3 + 2];
    float cx = nvs[(size_t)c * 3],  cy = nvs[(size_t)c * 3 + 1],  cz = nvs[(size_t)c * 3 + 2];
    float dot = vx * cx + vy * cy + vz * cz;
    float nrm = elen[eg] * (sqrtf(cx * cx + cy * cy + cz * cz) + 1e-6f);
    float ct = fminf(1.0f, fmaxf(-1.0f, dot / nrm));
    _Float16* rw = s_in + (tile0 + lane) * 320;
    rw[128] = (_Float16)ct;
#pragma unroll
    for (int j = 129; j < 160; ++j) rw[j] = (_Float16)0.0f;
  }

  // ---- Layer 3: 160 (padded 129) -> 128, swish -> t1 (reuses s_h1) ----
#pragma unroll
  for (int nt = 0; nt < 8; ++nt) {
    v8f acc = {};
#pragma unroll
    for (int kt = 0; kt < 5; ++kt) {
      v16h a = load_A_lds(s_in + tile0 * 320, 320, kt, lane);
      v16h b = load_B(wfrag, TW1_BASE + kt * 8 + nt, lane);
      acc = WMMA_F16(a, b, acc);
    }
    float bn = tb1[nt * 16 + n];
#pragma unroll
    for (int r = 0; r < 8; ++r)
      s_h1[(tile0 + rb + r) * 128 + nt * 16 + n] = (_Float16)swishf(acc[r] + bn);
  }

  // ---- Layer 4: 128 -> 128, sigmoid gate; scatter m*g into aggr ----
  int nodeId[8];
#pragma unroll
  for (int r = 0; r < 8; ++r) nodeId[r] = coli[e0 + rb + r];

#pragma unroll
  for (int nt = 0; nt < 8; ++nt) {
    v8f acc = {};
#pragma unroll
    for (int kt = 0; kt < 4; ++kt) {
      v16h a = load_A_lds(s_h1 + tile0 * 128, 128, kt, lane);
      v16h b = load_B(wfrag, TW2_BASE + kt * 8 + nt, lane);
      acc = WMMA_F16(a, b, acc);
    }
    float bn = tb2[nt * 16 + n];
#pragma unroll
    for (int r = 0; r < 8; ++r) {
      float g  = sigmf(acc[r] + bn);
      float mv = (float)s_in[(tile0 + rb + r) * 320 + nt * 16 + n];
      atomicAdd(aggr + (size_t)nodeId[r] * DIM + nt * 16 + n, mv * g);
    }
  }
}

// ---------------- node MLP: [x | aggr] 256 -> 128 -> 128, residual ------------
__global__ __launch_bounds__(128) void node_kernel(
    const float* __restrict__ x, const _Float16* __restrict__ wfrag,
    const float* __restrict__ nb1, const float* __restrict__ nb2,
    float* out /* in: aggr, out: x + x_new (aliased, rows are wave-private) */) {
  __shared__ _Float16 s_in[64 * 256];
  __shared__ _Float16 s_h1[64 * 128];

  int lane  = threadIdx.x & 31;
  int wave  = threadIdx.x >> 5;
  int tile0 = wave * 16;
  int n0    = blockIdx.x * 64 + tile0;
  int n     = lane & 15;
  int rb    = (lane >> 4) * 8;

#pragma unroll 1
  for (int i = 0; i < 16; ++i) {
    int nid = n0 + i; if (nid >= NNODES) nid = NNODES - 1;
    float4 xa = *(const float4*)(x   + (size_t)nid * DIM + lane * 4);
    float4 ag = *(const float4*)(out + (size_t)nid * DIM + lane * 4);
    _Float16* rw = s_in + (tile0 + i) * 256;
    v4h h;
    h[0] = (_Float16)xa.x; h[1] = (_Float16)xa.y; h[2] = (_Float16)xa.z; h[3] = (_Float16)xa.w;
    *(v4h*)(rw + lane * 4) = h;
    h[0] = (_Float16)ag.x; h[1] = (_Float16)ag.y; h[2] = (_Float16)ag.z; h[3] = (_Float16)ag.w;
    *(v4h*)(rw + 128 + lane * 4) = h;
  }

#pragma unroll
  for (int nt = 0; nt < 8; ++nt) {
    v8f acc = {};
#pragma unroll
    for (int kt = 0; kt < 8; ++kt) {
      v16h a = load_A_lds(s_in + tile0 * 256, 256, kt, lane);
      v16h b = load_B(wfrag, NW1_BASE + kt * 8 + nt, lane);
      acc = WMMA_F16(a, b, acc);
    }
    float bn = nb1[nt * 16 + n];
#pragma unroll
    for (int r = 0; r < 8; ++r)
      s_h1[(tile0 + rb + r) * 128 + nt * 16 + n] = (_Float16)swishf(acc[r] + bn);
  }

#pragma unroll
  for (int nt = 0; nt < 8; ++nt) {
    v8f acc = {};
#pragma unroll
    for (int kt = 0; kt < 4; ++kt) {
      v16h a = load_A_lds(s_h1 + tile0 * 128, 128, kt, lane);
      v16h b = load_B(wfrag, NW2_BASE + kt * 8 + nt, lane);
      acc = WMMA_F16(a, b, acc);
    }
    float bn = nb2[nt * 16 + n];
#pragma unroll
    for (int r = 0; r < 8; ++r) {
      int nid = n0 + rb + r;
      if (nid < NNODES) {
        float v = swishf(acc[r] + bn);
        out[(size_t)nid * DIM + nt * 16 + n] = x[(size_t)nid * DIM + nt * 16 + n] + v;
      }
    }
  }
}

extern "C" void kernel_launch(void* const* d_in, const int* in_sizes, int n_in,
                              void* d_out, int out_size, void* d_ws, size_t ws_size,
                              hipStream_t stream) {
  (void)in_sizes; (void)n_in; (void)out_size; (void)ws_size;
  const float* x     = (const float*)d_in[0];
  const int*   ei    = (const int*)d_in[1];
  const float* eattr = (const float*)d_in[2];
  const float* evec  = (const float*)d_in[3];
  const float* elen  = (const float*)d_in[4];
  const float* ew1   = (const float*)d_in[5];
  const float* eb1   = (const float*)d_in[6];
  const float* ew2   = (const float*)d_in[7];
  const float* eb2   = (const float*)d_in[8];
  const float* nw1   = (const float*)d_in[9];
  const float* nb1   = (const float*)d_in[10];
  const float* nw2   = (const float*)d_in[11];
  const float* nb2   = (const float*)d_in[12];
  const float* tw1   = (const float*)d_in[13];
  const float* tb1   = (const float*)d_in[14];
  const float* tw2   = (const float*)d_in[15];
  const float* tb2   = (const float*)d_in[16];

  float* out = (float*)d_out;
  char*  ws  = (char*)d_ws;
  _Float16* wfrag = (_Float16*)ws;
  float* vsum = (float*)(ws + VEC_OFF);
  float* cnt  = (float*)(ws + CNT_OFF);
  float* nvs  = (float*)(ws + NVS_OFF);
  const int* rowi = ei;
  const int* coli = ei + NEDGES;

  // zero aggr accumulator (d_out) and vec_sum+counts (contiguous region)
  hipMemsetAsync(out, 0, (size_t)NNODES * DIM * sizeof(float), stream);
  hipMemsetAsync(ws + VEC_OFF, 0, (size_t)NNODES * 4 * sizeof(float), stream);

  prep_weights<<<NFRAGS, 32, 0, stream>>>(ew1, ew2, tw1, tw2, nw1, nw2, wfrag);
  scatter_vec<<<(NEDGES + 255) / 256, 256, 0, stream>>>(coli, evec, vsum, cnt);
  normalize_vec<<<(NNODES + 255) / 256, 256, 0, stream>>>(vsum, cnt, nvs);
  edge_kernel<<<NEDGES / 64, 128, 0, stream>>>(x, rowi, coli, eattr, evec, elen, nvs,
                                               wfrag, eb1, eb2, tb1, tb2, out);
  node_kernel<<<(NNODES + 63) / 64, 128, 0, stream>>>(x, wfrag, nb1, nb2, out);
}
